// EGNN_AD2_CV_16312285790222
// MI455X (gfx1250) — matrix operand order repaired and checked by reading, not compiled, not run on hardware
//
#include <hip/hip_runtime.h>
#include <hip/hip_bf16.h>

// ---------------------------------------------------------------------------
// EGNN_AD2_CV fused forward for gfx1250 (MI455X).
// One workgroup (256 threads = 8 wave32) per graph of 22 nodes / 462 edges.
// All intermediates live in LDS; matrix math via v_wmma_f32_16x16x32_f16.
// Per-layer weights staged global->LDS via global_load_async_to_lds_b128.
// ---------------------------------------------------------------------------

#define NP      22          // particles per graph
#define HID     64
#define EB      462         // directed edges per graph (22*21)
#define EPAD    464         // padded to 29 tiles of 16
#define ETILES  29
#define NLAY    4

// packed weight tile map (per layer): 60 tiles of 512 halves (32 lanes x 16)
#define T_E1    0           // edge_w1 : 5 K-chunks x 4 N-tiles (K=130 pad 160)
#define T_E2    20          // edge_w2 : 2 x 4   (K=64)
#define T_CW    28          // coord_w1: 2 x 4   (K=64)
#define T_N1    36          // node_w1 : 4 x 4   (K=128)
#define T_N2    52          // node_w2 : 2 x 4   (K=64)
#define PACK_PER_LAYER (60*512)     // halves
#define PACK_TOTAL     (NLAY*PACK_PER_LAYER)

// bias table layout (floats in LDS)
#define BIAS_E1   0
#define BIAS_E2   64
#define BIAS_C1   128
#define OFF_CW2   192
#define BIAS_N1   256
#define BIAS_N2   320

typedef _Float16 v16h __attribute__((ext_vector_type(16)));
typedef _Float16 v8h  __attribute__((ext_vector_type(8)));
typedef float    v8f  __attribute__((ext_vector_type(8)));

union V16U { v16h v; v8h h[2]; };

__device__ __forceinline__ float siluf(float x) { return x / (1.f + __expf(-x)); }

// A fragment (16x32 f16): lane holds row M=lane&15; K = {half*8+0..7} U {16+half*8+0..7}
// caller passes p = rowptr + kchunk + half*8   -> two contiguous 16B chunks.
__device__ __forceinline__ v16h load_afrag(const _Float16* p) {
  V16U u;
  u.h[0] = *(const v8h*)(p);
  u.h[1] = *(const v8h*)(p + 16);
  return u.v;
}

// B fragment (32x16 f16), pre-packed: 16 contiguous halves per lane per tile.
__device__ __forceinline__ v16h load_bfrag(const _Float16* wblob, int tile, int lane) {
  const _Float16* p = wblob + ((tile * 32 + lane) << 4);
  V16U u;
  u.h[0] = *(const v8h*)(p);
  u.h[1] = *(const v8h*)(p + 8);
  return u.v;
}

__device__ __forceinline__ v8f wmma_f16(v16h a, v16h b, v8f c) {
  return __builtin_amdgcn_wmma_f32_16x16x32_f16(false, a, false, b, (short)0, c,
                                                false, false);
}

// ---------------------------------------------------------------------------
// Pack all layer weights into f16 B-fragment order.
// write index == gid:  ((layer*60 + tile)*32 + lane)*16 + t
// value: W[layer][k][n], k = kc*32 + (t<8 ? (lane>>4)*8 + t : 16 + (lane>>4)*8 + t-8), n = nt*16 + (lane&15)
// ---------------------------------------------------------------------------
__global__ __launch_bounds__(256) void egnn_pack_weights(
    const float* __restrict__ ew1, const float* __restrict__ ew2,
    const float* __restrict__ cw1, const float* __restrict__ nw1,
    const float* __restrict__ nw2, _Float16* __restrict__ packed) {
  int gid = blockIdx.x * 256 + threadIdx.x;
  if (gid >= PACK_TOTAL) return;
  int t16 = gid & 15;
  int L   = (gid >> 4) & 31;
  int tg  = (gid >> 9) % 60;
  int l   = (gid >> 9) / 60;
  int kc, nt, Kact;
  const float* src;
  if (tg < T_E2)      { int u = tg;        kc = u >> 2; nt = u & 3; Kact = 130; src = ew1 + l * 130 * 64; }
  else if (tg < T_CW) { int u = tg - T_E2; kc = u >> 2; nt = u & 3; Kact = 64;  src = ew2 + l * 64 * 64; }
  else if (tg < T_N1) { int u = tg - T_CW; kc = u >> 2; nt = u & 3; Kact = 64;  src = cw1 + l * 64 * 64; }
  else if (tg < T_N2) { int u = tg - T_N1; kc = u >> 2; nt = u & 3; Kact = 128; src = nw1 + l * 128 * 64; }
  else                { int u = tg - T_N2; kc = u >> 2; nt = u & 3; Kact = 64;  src = nw2 + l * 64 * 64; }
  int k = kc * 32 + ((L >> 4) << 3) + (t16 & 7) + ((t16 >> 3) << 4);
  int n = nt * 16 + (L & 15);
  float v = (k < Kact) ? src[k * 64 + n] : 0.f;
  packed[gid] = (_Float16)v;
}

// ---------------------------------------------------------------------------
// Fused EGNN forward: one block per graph.
// Dynamic LDS layout (bytes):
//   halves: wblob 30720 | h16 1408 | tmp 8*16*64 | hcat 32*128 | rad16 464 | ea16 464
//   floats: h32 1408 | agg 1408 | xcur 72 | x0 72 | diffn 1392 | scal 464 | bias 384
//   ints  : edgenode 464
// ---------------------------------------------------------------------------
#define SMEM_HALVES (30720 + 1408 + 8192 + 4096 + 464 + 464)
#define SMEM_BYTES  (SMEM_HALVES * 2 + (1408 + 1408 + 72 + 72 + 1392 + 464 + 384 + 464) * 4)

__global__ __launch_bounds__(256) void egnn_fused(
    const float* __restrict__ tin, const float* __restrict__ xs,
    const float* __restrict__ h_init, const float* __restrict__ emb_w,
    const float* __restrict__ emb_b,
    const float* __restrict__ edge_b1, const float* __restrict__ edge_b2,
    const float* __restrict__ node_b1, const float* __restrict__ node_b2,
    const float* __restrict__ coord_b1, const float* __restrict__ coord_w2,
    const _Float16* __restrict__ packed, float* __restrict__ out) {
  extern __shared__ char smem_raw[];
  _Float16* wblob = (_Float16*)smem_raw;           // 30720
  _Float16* h16   = wblob + 30720;                 // 1408
  _Float16* tmp   = h16 + 1408;                    // 8192 (per-wave 16x64 staging)
  _Float16* hcat  = tmp + 8192;                    // 4096 (32 x 128)
  _Float16* rad16 = hcat + 4096;                   // 464
  _Float16* ea16  = rad16 + 464;                   // 464
  float* fbase = (float*)(smem_raw + SMEM_HALVES * 2);
  float* h32   = fbase;                            // 1408
  float* agg   = h32 + 1408;                       // 1408
  float* xcur  = agg + 1408;                       // 72
  float* x0s   = xcur + 72;                        // 72
  float* diffn = x0s + 72;                         // 1392
  float* scal  = diffn + 464 * 3;                  // 464
  float* bias  = scal + 464;                       // 384
  int* edgenode = (int*)(bias + 384);              // 464: n | (c<<5)

  const int b    = blockIdx.x;
  const int tid  = threadIdx.x;
  const int lane = tid & 31;
  const int wave = tid >> 5;
  const int Ml   = lane & 15;
  const int half = lane >> 4;

  // ---- setup: coords + embedding + static edge tables -----------------
  const float tb = tin[b];
  for (int i = tid; i < NP * 3; i += 256) {
    float v = xs[b * (NP * 3) + i];
    x0s[i] = v;
    xcur[i] = v;
  }
  for (int i = tid; i < NP * HID; i += 256) {
    int n = i >> 6, c = i & 63;
    float acc = emb_b[c];
#pragma unroll
    for (int k = 0; k < 8; ++k) acc += h_init[n * 8 + k] * emb_w[k * 64 + c];
    acc += tb * emb_w[8 * 64 + c];
    h32[i] = acc;
    h16[i] = (_Float16)acc;
  }
  __syncthreads();
  for (int e = tid; e < EPAD; e += 256) {
    float v = 0.f;
    int n = 0, c = 0;
    if (e < EB) {
      n = e / 21;
      int jj = e - n * 21;
      c = jj + (jj >= n);
      float dx = x0s[n * 3 + 0] - x0s[c * 3 + 0];
      float dy = x0s[n * 3 + 1] - x0s[c * 3 + 1];
      float dz = x0s[n * 3 + 2] - x0s[c * 3 + 2];
      v = dx * dx + dy * dy + dz * dz;
    }
    ea16[e] = (_Float16)v;
    edgenode[e] = n | (c << 5);
  }

  const unsigned wblob_lds = (unsigned)(size_t)(void*)wblob;  // LDS byte offset

  for (int l = 0; l < NLAY; ++l) {
    // ---- phase A: async-stage layer weights; biases/diffn overlap ------
    {
      const unsigned long long gbase =
          (unsigned long long)(const void*)(packed + (size_t)l * PACK_PER_LAYER);
#pragma unroll
      for (int it = 0; it < 15; ++it) {            // 15*256*16B = 61440B
        unsigned goff = (unsigned)(tid + it * 256) * 16u;
        unsigned ldsa = wblob_lds + goff;
        asm volatile("global_load_async_to_lds_b128 %0, %1, %2"
                     :
                     : "v"(ldsa), "v"(goff), "s"(gbase)
                     : "memory");
      }
    }
    if (tid < 64) {
      bias[BIAS_E1 + tid] = edge_b1[l * 64 + tid];
      bias[BIAS_E2 + tid] = edge_b2[l * 64 + tid];
      bias[BIAS_C1 + tid] = coord_b1[l * 64 + tid];
      bias[OFF_CW2 + tid] = coord_w2[l * 64 + tid];
      bias[BIAS_N1 + tid] = node_b1[l * 64 + tid];
      bias[BIAS_N2 + tid] = node_b2[l * 64 + tid];
    }
    for (int i = tid; i < NP * HID; i += 256) agg[i] = 0.f;
    for (int e = tid; e < EPAD; e += 256) {
      int rc = edgenode[e];                         // pad rows -> n=c=0 -> zeros
      int n = rc & 31, c = rc >> 5;
      float dx = xcur[n * 3 + 0] - xcur[c * 3 + 0];
      float dy = xcur[n * 3 + 1] - xcur[c * 3 + 1];
      float dz = xcur[n * 3 + 2] - xcur[c * 3 + 2];
      float r = dx * dx + dy * dy + dz * dz;
      float inv = rsqrtf(r + 1e-8f);
      diffn[e * 3 + 0] = dx * inv;
      diffn[e * 3 + 1] = dy * inv;
      diffn[e * 3 + 2] = dz * inv;
      rad16[e] = (_Float16)r;
    }
    asm volatile("s_wait_asynccnt 0" ::: "memory");
    __syncthreads();

    // hoist layer-invariant B fragments for GEMM2 + coord MLP into VGPRs
    v16h bE2[8], bCW[8];
#pragma unroll
    for (int q = 0; q < 8; ++q) {
      bE2[q] = load_bfrag(wblob, T_E2 + q, lane);
      bCW[q] = load_bfrag(wblob, T_CW + q, lane);
    }

    // ---- phase B: edge tiles (16 edges per wave iteration) -------------
    _Float16* tw = tmp + wave * (16 * 64);
    for (int et = wave; et < ETILES; et += 8) {
      const int e0 = et * 16;
      const int e = e0 + Ml;
      const int rc = edgenode[e];
      const int rown = rc & 31, coln = rc >> 5;
      // GEMM1: ef(16x160) x edge_w1(160x64)
      v8f acc[4];
#pragma unroll
      for (int nt = 0; nt < 4; ++nt) { v8f z = {}; acc[nt] = z; }
#pragma unroll
      for (int kc = 0; kc < 5; ++kc) {
        v16h a;
        if (kc < 4) {
          const _Float16* rp = (kc < 2) ? (h16 + rown * 64 + kc * 32)
                                        : (h16 + coln * 64 + (kc - 2) * 32);
          a = load_afrag(rp + half * 8);
        } else {
          v16h z = {};
          if (lane < 16) { z[0] = rad16[e]; z[1] = ea16[e]; }
          a = z;
        }
#pragma unroll
        for (int nt = 0; nt < 4; ++nt)
          acc[nt] = wmma_f16(a, load_bfrag(wblob, T_E1 + kc * 4 + nt, lane), acc[nt]);
      }
#pragma unroll
      for (int nt = 0; nt < 4; ++nt) {
        const float bc = bias[BIAS_E1 + nt * 16 + Ml];
#pragma unroll
        for (int r = 0; r < 8; ++r)
          tw[(r + 8 * half) * 64 + nt * 16 + Ml] = (_Float16)siluf(acc[nt][r] + bc);
      }
      // GEMM2: (16x64) x edge_w2(64x64) -> messages m  (B from registers)
      v8f acc2[4];
#pragma unroll
      for (int nt = 0; nt < 4; ++nt) { v8f z = {}; acc2[nt] = z; }
#pragma unroll
      for (int kc = 0; kc < 2; ++kc) {
        v16h a = load_afrag(tw + Ml * 64 + kc * 32 + half * 8);
#pragma unroll
        for (int nt = 0; nt < 4; ++nt)
          acc2[nt] = wmma_f16(a, bE2[kc * 4 + nt], acc2[nt]);
      }
#pragma unroll
      for (int nt = 0; nt < 4; ++nt) {
        const float bc = bias[BIAS_E2 + nt * 16 + Ml];
#pragma unroll
        for (int r = 0; r < 8; ++r) {
          float mv = siluf(acc2[nt][r] + bc);
          tw[(r + 8 * half) * 64 + nt * 16 + Ml] = (_Float16)mv;  // stage m
          int er = e0 + r + 8 * half;
          if (er < EB) {
            int nn = edgenode[er] & 31;
            atomicAdd(&agg[nn * 64 + nt * 16 + Ml], mv);          // segment sum of m
          }
        }
      }
      // coord MLP: silu(m @ coord_w1 + b) @ coord_w2 -> scal per edge
      v8f acc3[4];
#pragma unroll
      for (int nt = 0; nt < 4; ++nt) { v8f z = {}; acc3[nt] = z; }
#pragma unroll
      for (int kc = 0; kc < 2; ++kc) {
        v16h a = load_afrag(tw + Ml * 64 + kc * 32 + half * 8);
#pragma unroll
        for (int nt = 0; nt < 4; ++nt)
          acc3[nt] = wmma_f16(a, bCW[kc * 4 + nt], acc3[nt]);
      }
      float pr[8];
#pragma unroll
      for (int r = 0; r < 8; ++r) pr[r] = 0.f;
#pragma unroll
      for (int nt = 0; nt < 4; ++nt) {
        const float bc = bias[BIAS_C1 + nt * 16 + Ml];
        const float w2 = bias[OFF_CW2 + nt * 16 + Ml];
#pragma unroll
        for (int r = 0; r < 8; ++r) pr[r] += siluf(acc3[nt][r] + bc) * w2;
      }
#pragma unroll
      for (int m = 1; m < 16; m <<= 1)
#pragma unroll
        for (int r = 0; r < 8; ++r) pr[r] += __shfl_xor(pr[r], m, 32);
      if (lane == 0) {
#pragma unroll
        for (int r = 0; r < 8; ++r)
          if (e0 + r < EB) scal[e0 + r] = pr[r];
      } else if (lane == 16) {
#pragma unroll
        for (int r = 0; r < 8; ++r)
          if (e0 + 8 + r < EB) scal[e0 + 8 + r] = pr[r];
      }
    }
    __syncthreads();

    // ---- phase C: coordinate update + node-MLP input [h | agg] ---------
    for (int i = tid; i < NP * 3; i += 256) {
      int n = i / 3, d = i - n * 3;
      float s = 0.f;
#pragma unroll
      for (int jj = 0; jj < 21; ++jj) {
        int e = n * 21 + jj;
        s += diffn[e * 3 + d] * scal[e];
      }
      xcur[i] += s;
    }
    for (int i = tid; i < 32 * 64; i += 256) {
      int n = i >> 6, c = i & 63;
      _Float16 hv = (_Float16)0.f;
      float av = 0.f;
      if (n < NP) { hv = h16[n * 64 + c]; av = agg[n * 64 + c]; }
      hcat[n * 128 + c] = hv;
      hcat[n * 128 + 64 + c] = (_Float16)av;
    }
    __syncthreads();

    // ---- phase D: node MLP + residual (2 row tiles on waves 0,1) -------
    if (wave < 2) {
      const int rowbase = wave * 16;
      const int g = rowbase + Ml;
      v8f acc[4];
#pragma unroll
      for (int nt = 0; nt < 4; ++nt) { v8f z = {}; acc[nt] = z; }
#pragma unroll
      for (int kc = 0; kc < 4; ++kc) {
        v16h a = load_afrag(hcat + g * 128 + kc * 32 + half * 8);
#pragma unroll
        for (int nt = 0; nt < 4; ++nt)
          acc[nt] = wmma_f16(a, load_bfrag(wblob, T_N1 + kc * 4 + nt, lane), acc[nt]);
      }
      _Float16* twn = tmp + wave * (16 * 64);
#pragma unroll
      for (int nt = 0; nt < 4; ++nt) {
        const float bc = bias[BIAS_N1 + nt * 16 + Ml];
#pragma unroll
        for (int r = 0; r < 8; ++r)
          twn[(r + 8 * half) * 64 + nt * 16 + Ml] = (_Float16)siluf(acc[nt][r] + bc);
      }
      v8f acc2[4];
#pragma unroll
      for (int nt = 0; nt < 4; ++nt) { v8f z = {}; acc2[nt] = z; }
#pragma unroll
      for (int kc = 0; kc < 2; ++kc) {
        v16h a = load_afrag(twn + Ml * 64 + kc * 32 + half * 8);
#pragma unroll
        for (int nt = 0; nt < 4; ++nt)
          acc2[nt] = wmma_f16(a, load_bfrag(wblob, T_N2 + kc * 4 + nt, lane), acc2[nt]);
      }
#pragma unroll
      for (int nt = 0; nt < 4; ++nt) {
        const float bc = bias[BIAS_N2 + nt * 16 + Ml];
#pragma unroll
        for (int r = 0; r < 8; ++r) {
          int gr = rowbase + r + 8 * half;
          if (gr < NP) {
            int idx = gr * 64 + nt * 16 + Ml;
            float nv = h32[idx] + acc2[nt][r] + bc;   // residual
            h32[idx] = nv;
            h16[idx] = (_Float16)nv;
          }
        }
      }
    }
    __syncthreads();
  }

  // ---- finalize: vel = (x - x0) minus per-dim mean ----------------------
  if (tid < 3) {
    float s = 0.f;
#pragma unroll
    for (int n = 0; n < NP; ++n) s += xcur[n * 3 + tid] - x0s[n * 3 + tid];
    scal[tid] = s * (1.f / (float)NP);
  }
  __syncthreads();
  for (int i = tid; i < NP * 3; i += 256) {
    int d = i % 3;
    out[(size_t)b * (NP * 3) + i] = (xcur[i] - x0s[i]) - scal[d];
  }
}

// ---------------------------------------------------------------------------
extern "C" void kernel_launch(void* const* d_in, const int* in_sizes, int n_in,
                              void* d_out, int out_size, void* d_ws, size_t ws_size,
                              hipStream_t stream) {
  const float* t        = (const float*)d_in[0];
  const float* xs       = (const float*)d_in[1];
  const float* h_init   = (const float*)d_in[2];
  const float* emb_w    = (const float*)d_in[3];
  const float* emb_b    = (const float*)d_in[4];
  // d_in[5], d_in[6]: emb_out_w/b — output of that matmul is unused by the reference.
  const float* edge_w1  = (const float*)d_in[7];
  const float* edge_b1  = (const float*)d_in[8];
  const float* edge_w2  = (const float*)d_in[9];
  const float* edge_b2  = (const float*)d_in[10];
  const float* node_w1  = (const float*)d_in[11];
  const float* node_b1  = (const float*)d_in[12];
  const float* node_w2  = (const float*)d_in[13];
  const float* node_b2  = (const float*)d_in[14];
  const float* coord_w1 = (const float*)d_in[15];
  const float* coord_b1 = (const float*)d_in[16];
  const float* coord_w2 = (const float*)d_in[17];
  float* out = (float*)d_out;

  _Float16* packed = (_Float16*)d_ws;   // PACK_TOTAL * 2 bytes = 245760
  const int Bn = in_sizes[0];           // batch count (t has B elements)

  egnn_pack_weights<<<(PACK_TOTAL + 255) / 256, 256, 0, stream>>>(
      edge_w1, edge_w2, coord_w1, node_w1, node_w2, packed);

  egnn_fused<<<Bn, 256, SMEM_BYTES, stream>>>(
      t, xs, h_init, emb_w, emb_b, edge_b1, edge_b2, node_b1, node_b2,
      coord_b1, coord_w2, packed, out);

  (void)n_in; (void)out_size; (void)ws_size;
}
